// PyTorch_FP8_MLP_23407571764131
// MI455X (gfx1250) — 8-bit matrix operand order repaired and checked by reading, compiled, not run on hardware
//
#include <hip/hip_runtime.h>
#include <hip/hip_bf16.h>


typedef __attribute__((ext_vector_type(16))) int   v16i;
typedef __attribute__((ext_vector_type(8)))  float v8f;
typedef int v4i_vs __attribute__((vector_size(16)));

#define BATCH 32768
#define DIN   784
#define DIN_P 896
#define DH    4096
#define DOUT  10
#define DOUT_P 16

#define LDS_STRIDE 144
#define A_BYTES   (128 * LDS_STRIDE)
#define B_BYTES   (256 * LDS_STRIDE)
#define BUF_BYTES (A_BYTES + B_BYTES)
#define SMEM_BYTES (2 * BUF_BYTES)

#if defined(__has_builtin)
#if __has_builtin(__builtin_amdgcn_cvt_pk_fp8_f32)
#define HW_FP8 1
#endif
#endif
#ifndef HW_FP8
#define HW_FP8 0
#endif

#if defined(__has_builtin)
#if __has_builtin(__builtin_amdgcn_global_load_async_to_lds_b128) && \
    __has_builtin(__builtin_amdgcn_s_wait_asynccnt)
#define HW_ASYNC 1
#endif
#endif
#ifndef HW_ASYNC
#define HW_ASYNC 0
#endif

#if HW_ASYNC
__device__ __forceinline__ void async_cp16(void* sp, const void* gp) {
    __builtin_amdgcn_global_load_async_to_lds_b128(
        (__attribute__((address_space(1))) v4i_vs*)(void*)gp,
        (__attribute__((address_space(3))) v4i_vs*)sp, 0, 0);
}
#endif

__device__ __forceinline__ unsigned char quant_e4m3_sw(float f) {
    unsigned int u    = __float_as_uint(f);
    unsigned int s    = (u >> 24) & 0x80u;
    unsigned int absu = u & 0x7FFFFFFFu;
    if (absu >= 0x7F800000u) return (unsigned char)(s | 0x7Fu);
    if (absu <  0x00800000u) return (unsigned char)s;
    int e = (int)(absu >> 23) - 127;
    unsigned int m = (absu & 0x7FFFFFu) | 0x800000u;
    int eb = e + 7;
    if (eb <= 0) {
        unsigned int shift = (unsigned int)(21 - eb);
        if (shift > 31u) return (unsigned char)s;
        unsigned int keep = m >> shift;
        unsigned int rem  = m & ((1u << shift) - 1u);
        unsigned int half = 1u << (shift - 1u);
        keep += (rem > half || (rem == half && (keep & 1u)));
        return (unsigned char)(s | keep);
    }
    unsigned int keep = m >> 20;
    unsigned int rem  = m & 0xFFFFFu;
    keep += (rem > 0x80000u || (rem == 0x80000u && (keep & 1u)));
    if (keep == 16u) { keep = 8u; eb++; }
    unsigned int code = ((unsigned int)eb << 3) | (keep & 7u);
    if (code >= 0x7Fu) code = 0x7Eu;
    return (unsigned char)(s | code);
}

__device__ __forceinline__ unsigned char quant1_e4m3(float f) {
#if HW_FP8
    return (unsigned char)(__builtin_amdgcn_cvt_pk_fp8_f32(f, 0.0f, 0, false) & 0xFF);
#else
    return quant_e4m3_sw(f);
#endif
}

__device__ __forceinline__ unsigned int quant4_e4m3(float a, float b, float c, float d) {
#if HW_FP8
    int v = 0;
    v = __builtin_amdgcn_cvt_pk_fp8_f32(a, b, v, false);
    v = __builtin_amdgcn_cvt_pk_fp8_f32(c, d, v, true);
    return (unsigned int)v;
#else
    return (unsigned int)quant_e4m3_sw(a)
         | ((unsigned int)quant_e4m3_sw(b) << 8)
         | ((unsigned int)quant_e4m3_sw(c) << 16)
         | ((unsigned int)quant_e4m3_sw(d) << 24);
#endif
}

__global__ __launch_bounds__(256) void quant_pad_kernel(
    const float* __restrict__ src, unsigned char* __restrict__ dst,
    int rows_out, int rows_in, int cols_in, int cols_out) {
    int idx = blockIdx.x * blockDim.x + threadIdx.x;
    int cpw = cols_out >> 2;
    int total = rows_out * cpw;
    if (idx >= total) return;
    int row = idx / cpw;
    int c0  = (idx - row * cpw) << 2;
    float f[4];
#pragma unroll
    for (int t = 0; t < 4; ++t) {
        int c = c0 + t;
        f[t] = (row < rows_in && c < cols_in) ? src[(size_t)row * cols_in + c] : 0.0f;
    }
    *(unsigned int*)(dst + (size_t)row * cols_out + c0) = quant4_e4m3(f[0], f[1], f[2], f[3]);
}

__device__ __forceinline__ v16i load_a_frag(const unsigned char* __restrict__ A,
                                            int kpad, int row_base, int kk, int lane) {
    int r = lane & 15, h = lane >> 4;
    const unsigned char* base = A + (size_t)(row_base + r) * kpad + kk + 16 * h;
    v16i f;
#pragma unroll
    for (int j = 0; j < 8; ++j) {
        uint2 d = *reinterpret_cast<const uint2*>(base + 32 * (j >> 1) + 8 * (j & 1));
        f[2 * j]     = (int)d.x;
        f[2 * j + 1] = (int)d.y;
    }
    return f;
}
__device__ __forceinline__ v16i load_b_frag(const unsigned char* __restrict__ W,
                                            int kpad, int col_base, int kk, int lane) {
    int n = lane & 15, h = lane >> 4;
    const unsigned char* base = W + (size_t)(col_base + n) * kpad + kk + 16 * h;
    v16i f;
#pragma unroll
    for (int j = 0; j < 4; ++j) {
        uint4 d = *reinterpret_cast<const uint4*>(base + 32 * j);
        f[4 * j] = (int)d.x; f[4 * j + 1] = (int)d.y;
        f[4 * j + 2] = (int)d.z; f[4 * j + 3] = (int)d.w;
    }
    return f;
}

__device__ __forceinline__ v16i lds_a_frag(const unsigned char* s, int row_base, int lane) {
    int r = lane & 15, h = lane >> 4;
    const unsigned char* base = s + (row_base + r) * LDS_STRIDE + 16 * h;
    v16i f;
#pragma unroll
    for (int j = 0; j < 8; ++j) {
        uint2 d = *reinterpret_cast<const uint2*>(base + 32 * (j >> 1) + 8 * (j & 1));
        f[2 * j]     = (int)d.x;
        f[2 * j + 1] = (int)d.y;
    }
    return f;
}
__device__ __forceinline__ v16i lds_b_frag(const unsigned char* s, int col_base, int lane) {
    int n = lane & 15, h = lane >> 4;
    const unsigned char* base = s + (col_base + n) * LDS_STRIDE + 16 * h;
    v16i f;
#pragma unroll
    for (int j = 0; j < 4; ++j) {
        uint4 d = *reinterpret_cast<const uint4*>(base + 32 * j);
        f[4 * j] = (int)d.x; f[4 * j + 1] = (int)d.y;
        f[4 * j + 2] = (int)d.z; f[4 * j + 3] = (int)d.w;
    }
    return f;
}

__global__ __launch_bounds__(512) void fc1_kernel(
    const unsigned char* __restrict__ xq,
    const unsigned char* __restrict__ w1q,
    unsigned char* __restrict__ hq) {
    extern __shared__ __align__(16) unsigned char smem[];

    int t    = threadIdx.x;
    int lane = t & 31;
    int wave = t >> 5;
    int wm   = (wave & 3) * 32;
    int wn   = (wave >> 2) * 64;
    int m0   = blockIdx.y * 128;
    int n0   = blockIdx.x * 256;

    int arow = t >> 2, aoff = (t & 3) * 32;
    int brow = t >> 1, boff = (t & 1) * 64;
    const unsigned char* agp = xq  + (size_t)(m0 + arow) * DIN_P + aoff;
    const unsigned char* bgp = w1q + (size_t)(n0 + brow) * DIN_P + boff;
    int a_l = arow * LDS_STRIDE + aoff;
    int b_l = brow * LDS_STRIDE + boff;

    v8f acc[2][4] = {};

#if HW_ASYNC
    {
        unsigned char* sA = smem;
        unsigned char* sB = smem + A_BYTES;
#pragma unroll
        for (int i = 0; i < 2; ++i) async_cp16(sA + a_l + 16 * i, agp + 16 * i);
#pragma unroll
        for (int i = 0; i < 4; ++i) async_cp16(sB + b_l + 16 * i, bgp + 16 * i);
    }
    int cur = 0;
    for (int kk = 0; kk < DIN_P; kk += 128) {
        __builtin_amdgcn_s_wait_asynccnt(0);
        __syncthreads();
        if (kk + 128 < DIN_P) {
            unsigned char* sA = smem + (cur ^ 1) * BUF_BYTES;
            unsigned char* sB = sA + A_BYTES;
#pragma unroll
            for (int i = 0; i < 2; ++i) async_cp16(sA + a_l + 16 * i, agp + kk + 128 + 16 * i);
#pragma unroll
            for (int i = 0; i < 4; ++i) async_cp16(sB + b_l + 16 * i, bgp + kk + 128 + 16 * i);
        }
        const unsigned char* sA = smem + cur * BUF_BYTES;
        const unsigned char* sB = sA + A_BYTES;
        v16i a0 = lds_a_frag(sA, wm,      lane);
        v16i a1 = lds_a_frag(sA, wm + 16, lane);
        v16i b[4];
#pragma unroll
        for (int j = 0; j < 4; ++j) b[j] = lds_b_frag(sB, wn + 16 * j, lane);
#pragma unroll
        for (int j = 0; j < 4; ++j) {
            acc[0][j] = __builtin_amdgcn_wmma_f32_16x16x128_fp8_fp8(
                a0, b[j], (short)0, acc[0][j], false, false);
            acc[1][j] = __builtin_amdgcn_wmma_f32_16x16x128_fp8_fp8(
                a1, b[j], (short)0, acc[1][j], false, false);
        }
        cur ^= 1;
    }
#else
    {
        unsigned char* sA = smem;
        unsigned char* sB = smem + A_BYTES;
        for (int kk = 0; kk < DIN_P; kk += 128) {
            uint4 ar[2], br[4];
#pragma unroll
            for (int i = 0; i < 2; ++i)
                ar[i] = *reinterpret_cast<const uint4*>(agp + kk + 16 * i);
#pragma unroll
            for (int i = 0; i < 4; ++i)
                br[i] = *reinterpret_cast<const uint4*>(bgp + kk + 16 * i);
            __syncthreads();
#pragma unroll
            for (int i = 0; i < 2; ++i)
                *reinterpret_cast<uint4*>(sA + a_l + 16 * i) = ar[i];
#pragma unroll
            for (int i = 0; i < 4; ++i)
                *reinterpret_cast<uint4*>(sB + b_l + 16 * i) = br[i];
            __syncthreads();
            v16i a0 = lds_a_frag(sA, wm,      lane);
            v16i a1 = lds_a_frag(sA, wm + 16, lane);
#pragma unroll
            for (int j = 0; j < 4; ++j) {
                v16i b = lds_b_frag(sB, wn + 16 * j, lane);
                acc[0][j] = __builtin_amdgcn_wmma_f32_16x16x128_fp8_fp8(
                    a0, b, (short)0, acc[0][j], false, false);
                acc[1][j] = __builtin_amdgcn_wmma_f32_16x16x128_fp8_fp8(
                    a1, b, (short)0, acc[1][j], false, false);
            }
        }
    }
#endif

    int coll = lane & 15, half = lane >> 4;
#pragma unroll
    for (int i = 0; i < 2; ++i) {
#pragma unroll
        for (int j = 0; j < 4; ++j) {
            int n = n0 + wn + 16 * j + coll;
#pragma unroll
            for (int v = 0; v < 8; ++v) {
                float val = acc[i][j][v];
                val = val > 0.0f ? val : 0.0f;
                int row = m0 + wm + 16 * i + v + 8 * half;
                hq[(size_t)row * DH + n] = quant1_e4m3(val);
            }
        }
    }
}

__global__ __launch_bounds__(256) void fc2_kernel(
    const unsigned char* __restrict__ hq,
    const unsigned char* __restrict__ w2q,
    float* __restrict__ out) {
    int lane = threadIdx.x & 31;
    int wave = threadIdx.x >> 5;
    int m0 = blockIdx.x * 256 + wave * 32;

    v8f acc[2] = {};

    for (int kk = 0; kk < DH; kk += 128) {
        v16i b  = load_b_frag(w2q, DH, 0, kk, lane);
        v16i a0 = load_a_frag(hq, DH, m0,      kk, lane);
        v16i a1 = load_a_frag(hq, DH, m0 + 16, kk, lane);
        acc[0] = __builtin_amdgcn_wmma_f32_16x16x128_fp8_fp8(
            a0, b, (short)0, acc[0], false, false);
        acc[1] = __builtin_amdgcn_wmma_f32_16x16x128_fp8_fp8(
            a1, b, (short)0, acc[1], false, false);
    }

    int col = lane & 15, half = lane >> 4;
    if (col < DOUT) {
#pragma unroll
        for (int i = 0; i < 2; ++i) {
#pragma unroll
            for (int v = 0; v < 8; ++v) {
                int row = m0 + 16 * i + v + 8 * half;
                out[(size_t)row * DOUT + col] = acc[i][v];
            }
        }
    }
}

extern "C" void kernel_launch(void* const* d_in, const int* in_sizes, int n_in,
                              void* d_out, int out_size, void* d_ws, size_t ws_size,
                              hipStream_t stream) {
    const float* x  = (const float*)d_in[0];
    const float* w1 = (const float*)d_in[1];
    const float* w2 = (const float*)d_in[2];
    float* out = (float*)d_out;

    unsigned char* ws  = (unsigned char*)d_ws;
    unsigned char* xq  = ws;
    unsigned char* w1q = xq  + (size_t)BATCH * DIN_P;
    unsigned char* w2q = w1q + (size_t)DH * DIN_P;
    unsigned char* hq  = w2q + (size_t)DOUT_P * DH;
    (void)in_sizes; (void)n_in; (void)out_size; (void)ws_size;

    {
        int total = BATCH * (DIN_P / 4);
        quant_pad_kernel<<<(total + 255) / 256, 256, 0, stream>>>(
            x, xq, BATCH, BATCH, DIN, DIN_P);
    }
    {
        int total = DH * (DIN_P / 4);
        quant_pad_kernel<<<(total + 255) / 256, 256, 0, stream>>>(
            w1, w1q, DH, DH, DIN, DIN_P);
    }
    {
        int total = DOUT_P * (DH / 4);
        quant_pad_kernel<<<(total + 255) / 256, 256, 0, stream>>>(
            w2, w2q, DOUT_P, DOUT, DH, DH);
    }

    fc1_kernel<<<dim3(DH / 256, BATCH / 128), 512, SMEM_BYTES, stream>>>(xq, w1q, hq);

    fc2_kernel<<<BATCH / 256, 256, 0, stream>>>(hq, w2q, out);
}
